// GCNNDoubleQCritic_58583353917872
// MI455X (gfx1250) — matrix operand order repaired and checked by reading, not compiled, not run on hardware
//
#include <hip/hip_runtime.h>

// ---------------------------------------------------------------------------
// GCNNDoubleQCritic on gfx1250 (CDNA5) — weight-stationary bf16 WMMA design.
//
// Reference collapses (A_hat = ones(3,3)/3, mean commutes with linear):
//   xbar = mean over 3 nodes of 36-dim node input           [B,36]
//   y0   = relu(xbar @ W0 + b0)                             [B,256]
//   y1   = relu(y0  @ W1 + b1)                              [B,256]
//   q    = y1 @ W2 + b2  (broadcast to 3 nodes)             [B,3]
//
// Block = 512 threads = 16 waves; wave w is stationary owner of output
// columns [w*16, w*16+16): its W0/W1 B-fragments live in VGPRs for the whole
// kernel. Block loops over 4 chunks of 64 rows; activations stage through LDS
// as bf16. Final GEMV folds into the GEMM2 epilogue via shfl_xor + ds_add_f32.
// ---------------------------------------------------------------------------

#define BS_TOTAL   131072
#define HID        256
#define CHUNKROWS  64
#define CHUNKS     4
#define GRIDX      (BS_TOTAL / (CHUNKROWS * CHUNKS))   // 512
#define THREADS    512

typedef __attribute__((ext_vector_type(16))) __bf16 v16bf;
typedef __attribute__((ext_vector_type(8)))  float  v8f;

struct alignas(16) U4 { unsigned x, y, z, w; };

union Frag {
    v16bf     v;
    unsigned  w[8];
    U4        q[2];
};

// ---------------------------------------------------------------------------
// Prologue: fp32 weights [Ksrc x N] -> bf16 WMMA B-fragments (lane-major pack:
// dword index in fragment = lane*8 + vgpr -> two global_load_b128 per lane).
// B pack used here (contiguous k, 32x16): lanes 0-15 K=0..15 (2/VGPR), lanes 16-31 K=16..31,
// column N = lane & 15. K zero-padded to a multiple of 32.
// ---------------------------------------------------------------------------
__global__ void convert_b_frags(const float* __restrict__ src,
                                unsigned int* __restrict__ dst,
                                int N, int Ksrc, int nS, int totalWords) {
    int t = blockIdx.x * blockDim.x + threadIdx.x;
    if (t >= totalWords) return;
    int word  = t & 255;
    int frag  = t >> 8;
    int s     = frag % nS;
    int j     = frag / nS;
    int lane  = word >> 3;
    int v     = word & 7;
    int n     = j * 16 + (lane & 15);
    int khalf = lane >> 4;
    int k0    = s * 32 + khalf * 16 + 2 * v;
    float f0  = (k0     < Ksrc) ? src[(size_t)k0       * N + n] : 0.0f;
    float f1  = (k0 + 1 < Ksrc) ? src[(size_t)(k0 + 1) * N + n] : 0.0f;
    union { __bf16 h[2]; unsigned u; } p;
    p.h[0] = (__bf16)f0;
    p.h[1] = (__bf16)f1;
    dst[(size_t)frag * 256 + word] = p.u;
}

// ---------------------------------------------------------------------------
// Main fused kernel. grid = (GRIDX, 2 heads), block = 512 threads (16 waves).
// ws dword offsets: W0 frags: head*8192 ; W1 frags: 16384 + head*32768.
// ---------------------------------------------------------------------------
__global__ __launch_bounds__(THREADS)
void gcnn_dq_kernel(const float* __restrict__ obs,
                    const float* __restrict__ act,
                    const unsigned int* __restrict__ ws,
                    const float* __restrict__ b0a, const float* __restrict__ b0b,
                    const float* __restrict__ b1a, const float* __restrict__ b1b,
                    const float* __restrict__ W2a, const float* __restrict__ W2b,
                    const float* __restrict__ b2a, const float* __restrict__ b2b,
                    float* __restrict__ out) {
    const int head = blockIdx.y;

    const float* b0p = head ? b0b : b0a;
    const float* b1p = head ? b1b : b1a;
    const float* W2p = head ? W2b : W2a;
    const float* b2p = head ? b2b : b2a;
    const unsigned int* W0f = ws + (size_t)head * 8192;
    const unsigned int* W1f = ws + 16384 + (size_t)head * 32768;

    __shared__ __bf16 xbar[CHUNKROWS][64];     // K padded 36 -> 64   (8 KB)
    __shared__ __bf16 y0[CHUNKROWS][HID];      // hidden activations (32 KB)
    __shared__ float  dotAcc[CHUNKROWS];       // final GEMV accumulator

    const int tid   = threadIdx.x;
    const int lane  = tid & 31;
    const int wv    = tid >> 5;          // wave id == stationary N-tile j
    const int khalf = lane >> 4;
    const int nl    = lane & 15;

    // ---- Stationary state: this wave's B fragments + epilogue scalars ----
    Frag B0[2], B1[8];
#pragma unroll
    for (int s = 0; s < 2; ++s) {
        const U4* bp = (const U4*)(W0f + (size_t)(wv * 2 + s) * 256) + lane * 2;
        B0[s].q[0] = bp[0]; B0[s].q[1] = bp[1];
    }
#pragma unroll
    for (int s = 0; s < 8; ++s) {
        const U4* bp = (const U4*)(W1f + (size_t)(wv * 8 + s) * 256) + lane * 2;
        B1[s].q[0] = bp[0]; B1[s].q[1] = bp[1];
    }
    const float bias0 = b0p[wv * 16 + nl];
    const float bias1 = b1p[wv * 16 + nl];
    const float w2v   = W2p[wv * 16 + nl];
    const float b2v   = b2p[0];

    for (int c = 0; c < CHUNKS; ++c) {
        const int rowBase = (blockIdx.x * CHUNKS + c) * CHUNKROWS;

        // ---- Stage 0: node-mean aggregation -> xbar (bf16, K zero-pad) ----
        for (int e = tid; e < CHUNKROWS * 64; e += THREADS) {
            int r = e >> 6, k = e & 63;
            size_t gr = (size_t)(rowBase + r);
            float val = 0.0f;
            if (k < 32) {
                const float* p = obs + gr * 96 + k;
                val = (p[0] + p[32] + p[64]) * (1.0f / 3.0f);
            } else if (k < 36) {
                const float* p = act + gr * 12 + (k - 32);
                val = (p[0] + p[4] + p[8]) * (1.0f / 3.0f);
            }
            ((__bf16*)xbar)[e] = (__bf16)val;
        }
        if (tid < CHUNKROWS) dotAcc[tid] = b2v;
        if (c + 1 < CHUNKS)   // pull next chunk's obs toward L2/WGP$
            __builtin_prefetch(obs + (size_t)(rowBase + CHUNKROWS) * 96 + tid * 12, 0, 1);
        __syncthreads();

        // ---- Stage 1: y0 = relu(xbar @ W0 + b0), K = 64 ----
        const unsigned int* xbw = (const unsigned int*)xbar;
        for (int rt = 0; rt < CHUNKROWS / 16; ++rt) {
            const int mA = rt * 16 + nl;
            v8f acc = {0.f, 0.f, 0.f, 0.f, 0.f, 0.f, 0.f, 0.f};
#pragma unroll
            for (int s = 0; s < 2; ++s) {
                Frag af;
#pragma unroll
                for (int v = 0; v < 8; ++v) {
                    int kk = s * 32 + khalf * 16 + 2 * v;
                    af.w[v] = xbw[mA * 32 + (kk >> 1)];
                }
                acc = __builtin_amdgcn_wmma_f32_16x16x32_bf16(
                    false, af.v, false, B0[s].v, (short)0, acc, false, false);
            }
#pragma unroll
            for (int v = 0; v < 8; ++v) {
                float z = acc[v] + bias0;
                z = z > 0.0f ? z : 0.0f;
                y0[rt * 16 + khalf * 8 + v][wv * 16 + nl] = (__bf16)z;
            }
        }
        __syncthreads();

        // ---- Stage 2: y1 = relu(y0 @ W1 + b1), fused 256->1 dot ----
        const unsigned int* y0w = (const unsigned int*)y0;
        for (int rt = 0; rt < CHUNKROWS / 16; ++rt) {
            const int mA = rt * 16 + nl;
            v8f acc = {0.f, 0.f, 0.f, 0.f, 0.f, 0.f, 0.f, 0.f};
#pragma unroll
            for (int s = 0; s < 8; ++s) {
                Frag af;
#pragma unroll
                for (int v = 0; v < 8; ++v) {
                    int kk = s * 32 + khalf * 16 + 2 * v;
                    af.w[v] = y0w[mA * 128 + (kk >> 1)];
                }
                acc = __builtin_amdgcn_wmma_f32_16x16x32_bf16(
                    false, af.v, false, B1[s].v, (short)0, acc, false, false);
            }
#pragma unroll
            for (int v = 0; v < 8; ++v) {
                float z = acc[v] + bias1;
                z = z > 0.0f ? z : 0.0f;
                float d = z * w2v;                 // partial over 16 columns
                d += __shfl_xor(d, 1);
                d += __shfl_xor(d, 2);
                d += __shfl_xor(d, 4);
                d += __shfl_xor(d, 8);
                if (nl == 0)                       // lanes 0 / 16
                    atomicAdd(&dotAcc[rt * 16 + khalf * 8 + v], d);
            }
        }
        __syncthreads();

        // ---- Stage 3: emit q broadcast across the 3 nodes ----
        if (tid < CHUNKROWS) {
            float q = dotAcc[tid];
            float* o = out + (size_t)head * BS_TOTAL * 3
                           + (size_t)(rowBase + tid) * 3;
            o[0] = q; o[1] = q; o[2] = q;
        }
        __syncthreads();   // xbar/dotAcc reused next chunk
    }
}

// ---------------------------------------------------------------------------
extern "C" void kernel_launch(void* const* d_in, const int* in_sizes, int n_in,
                              void* d_out, int out_size, void* d_ws, size_t ws_size,
                              hipStream_t stream) {
    (void)in_sizes; (void)n_in; (void)out_size; (void)ws_size;

    const float* obs   = (const float*)d_in[0];
    const float* actn  = (const float*)d_in[1];
    const float* q1_W0 = (const float*)d_in[2];
    const float* q1_b0 = (const float*)d_in[3];
    const float* q1_W1 = (const float*)d_in[4];
    const float* q1_b1 = (const float*)d_in[5];
    const float* q1_W2 = (const float*)d_in[6];
    const float* q1_b2 = (const float*)d_in[7];
    const float* q2_W0 = (const float*)d_in[8];
    const float* q2_b0 = (const float*)d_in[9];
    const float* q2_W1 = (const float*)d_in[10];
    const float* q2_b1 = (const float*)d_in[11];
    const float* q2_W2 = (const float*)d_in[12];
    const float* q2_b2 = (const float*)d_in[13];

    unsigned int* ws = (unsigned int*)d_ws;
    float* out = (float*)d_out;

    // Pre-swizzle weights to bf16 WMMA B-fragments in scratch (320 KB total).
    convert_b_frags<<<32,  256, 0, stream>>>(q1_W0, ws +     0, HID,  36, 2,  8192);
    convert_b_frags<<<32,  256, 0, stream>>>(q2_W0, ws +  8192, HID,  36, 2,  8192);
    convert_b_frags<<<128, 256, 0, stream>>>(q1_W1, ws + 16384, HID, 256, 8, 32768);
    convert_b_frags<<<128, 256, 0, stream>>>(q2_W1, ws + 49152, HID, 256, 8, 32768);

    dim3 grid(GRIDX, 2);
    gcnn_dq_kernel<<<grid, THREADS, 0, stream>>>(
        obs, actn, ws,
        q1_b0, q2_b0, q1_b1, q2_b1, q1_W2, q2_W2, q1_b2, q2_b2,
        out);
}
